// Encoder_27650999451954
// MI455X (gfx1250) — 8-bit matrix operand order repaired and checked by reading, compiled, not run on hardware
//
#include <hip/hip_runtime.h>
#include <hip/hip_bf16.h>
#include <stdint.h>


#define NHEADS 16
#define HDIM   128
#define QMUL   0.08838834764831843f

typedef int v8i __attribute__((ext_vector_type(8)));
typedef unsigned v4u __attribute__((ext_vector_type(4)));
typedef int v8s __attribute__((ext_vector_type(8)));
typedef int v4i_vs __attribute__((vector_size(16)));

#if defined(__has_builtin)
#if __has_builtin(__builtin_amdgcn_global_load_async_to_lds_b128)
#define USE_ASYNC_LDS 1
#endif
#endif
#ifndef USE_ASYNC_LDS
#define USE_ASYNC_LDS 0
#endif

#if USE_ASYNC_LDS
__device__ __forceinline__ void cp_async16(signed char* lds_dst, const signed char* gsrc) {
    __builtin_amdgcn_global_load_async_to_lds_b128(
        (__attribute__((address_space(1))) v4i_vs*)(uintptr_t)gsrc,
        (__attribute__((address_space(3))) v4i_vs*)(unsigned)(uintptr_t)lds_dst,
        0, 0);
}
__device__ __forceinline__ void wait_async0() {
#if __has_builtin(__builtin_amdgcn_s_wait_asynccnt)
    __builtin_amdgcn_s_wait_asynccnt(0);
#else
    asm volatile("s_wait_asynccnt 0x0" ::: "memory");
#endif
}
#endif

__device__ __forceinline__ float scale_from(const unsigned* bits, float premul) {
    float am = __uint_as_float(*bits);
    return fmaxf(am * premul * (1.0f / 127.0f), 1e-8f);
}

__device__ __forceinline__ int clamp127(int v) {
    return v < -127 ? -127 : (v > 127 ? 127 : v);
}

__global__ void tdm_probe_kernel(const signed char* src, signed char* dst, int width) {
    __shared__ __align__(128) signed char tile[64 * 64];
    const unsigned lds = (unsigned)(uintptr_t)&tile[0];
    const unsigned long long ga = (unsigned long long)(uintptr_t)src;
    v4u g0;
    g0.x = 1u;
    g0.y = lds;
    g0.z = (unsigned)ga;
    g0.w = (unsigned)((ga >> 32) & 0x01FFFFFFull) | (2u << 30);
    const unsigned td0 = (unsigned)width;
    const unsigned td1 = 64u;
    const unsigned long long st0 = (unsigned long long)width;
    v8s g1;
    g1[0] = 0;
    g1[1] = (int)((td0 & 0xFFFFu) << 16);
    g1[2] = (int)(((td0 >> 16) & 0xFFFFu) | ((td1 & 0xFFFFu) << 16));
    g1[3] = (int)(((td1 >> 16) & 0xFFFFu) | (64u << 16));
    g1[4] = (int)64u;
    g1[5] = (int)(st0 & 0xFFFFFFFFull);
    g1[6] = (int)((st0 >> 32) & 0xFFFFull);
    g1[7] = 0;
    asm volatile("tensor_load_to_lds %0, %1" :: "s"(g0), "s"(g1) : "memory");
    __builtin_amdgcn_s_wait_tensorcnt(0);
    __syncthreads();
    dst[threadIdx.x] = tile[threadIdx.x];
}

__global__ void init_scales(unsigned* U) {
    int i = threadIdx.x;
    if (i < 16) U[i] = (i == 8) ? 0x7F7FFFFFu : 0u;
}

__global__ __launch_bounds__(256) void absmax_k(const float* __restrict__ x,
                                                size_t n4, unsigned* __restrict__ out) {
    float m = 0.0f;
    const float4* p = reinterpret_cast<const float4*>(x);
    for (size_t i = (size_t)blockIdx.x * blockDim.x + threadIdx.x; i < n4;
         i += (size_t)gridDim.x * blockDim.x) {
        float4 v = p[i];
        m = fmaxf(m, fmaxf(fmaxf(fabsf(v.x), fabsf(v.y)),
                           fmaxf(fabsf(v.z), fabsf(v.w))));
    }
    for (int off = 16; off > 0; off >>= 1) m = fmaxf(m, __shfl_xor(m, off, 32));
    __shared__ float sm[8];
    if ((threadIdx.x & 31) == 0) sm[threadIdx.x >> 5] = m;
    __syncthreads();
    if (threadIdx.x == 0) {
        float mm = sm[0];
        for (int i = 1; i < 8; ++i) mm = fmaxf(mm, sm[i]);
        atomicMax(out, __float_as_uint(mm));
    }
}

__global__ __launch_bounds__(256) void quant_k(const float* __restrict__ x,
                                               signed char* __restrict__ q, size_t n,
                                               const unsigned* __restrict__ bits,
                                               float premul) {
    size_t i = (size_t)blockIdx.x * blockDim.x + threadIdx.x;
    if (i >= n) return;
    float s = scale_from(bits, premul);
    q[i] = (signed char)clamp127((int)rintf(x[i] * premul / s));
}

__global__ __launch_bounds__(256) void quantT_v(const float* __restrict__ vf,
                                                signed char* __restrict__ vT, size_t n,
                                                const unsigned* __restrict__ bits, int S) {
    size_t i = (size_t)blockIdx.x * blockDim.x + threadIdx.x;
    if (i >= n) return;
    float s = scale_from(bits, 1.0f);
    int d = (int)(i % HDIM);
    size_t t = i / HDIM;
    int h = (int)(t % NHEADS);
    size_t tok = t / NHEADS;
    int ss = (int)(tok % (size_t)S);
    int b  = (int)(tok / (size_t)S);
    int v = clamp127((int)rintf(vf[i] / s));
    vT[(((size_t)b * NHEADS + h) * HDIM + d) * (size_t)S + ss] = (signed char)v;
}

__global__ __launch_bounds__(256) void gemm_i8(const signed char* __restrict__ A,
                                               const signed char* __restrict__ Bw,
                                               const float* __restrict__ bias,
                                               float* __restrict__ C,
                                               int M, int N, int K,
                                               const unsigned* __restrict__ sa_bits,
                                               float sa_premul,
                                               const unsigned* __restrict__ sb_bits) {
    constexpr int AST = 80;
    constexpr int BST = 96;
    __shared__ __align__(16) signed char As[2][64 * AST];
    __shared__ __align__(16) signed char Bs[2][128 * BST];
    const int tid = threadIdx.x;
    const int wave = tid >> 5, lane = tid & 31;
    const int m_sub = wave >> 1, n_sub = wave & 1;
    const int m0 = blockIdx.y * 64, n0 = blockIdx.x * 128;
    const int grp = lane >> 4, ln = lane & 15;
    const int lr = tid >> 2, lc = tid & 3;

    auto stage = [&](int buf, int kc) {
        const signed char* ga  = A  + (size_t)(m0 + lr) * K + kc + lc * 16;
        const signed char* gb0 = Bw + (size_t)(n0 + lr) * K + kc + lc * 16;
        const signed char* gb1 = Bw + (size_t)(n0 + 64 + lr) * K + kc + lc * 16;
        signed char* la  = &As[buf][lr * AST + lc * 16];
        signed char* lb0 = &Bs[buf][lr * BST + lc * 16];
        signed char* lb1 = &Bs[buf][(64 + lr) * BST + lc * 16];
#if USE_ASYNC_LDS
        cp_async16(la, ga);
        cp_async16(lb0, gb0);
        cp_async16(lb1, gb1);
#else
        *reinterpret_cast<uint4*>(la)  = *reinterpret_cast<const uint4*>(ga);
        *reinterpret_cast<uint4*>(lb0) = *reinterpret_cast<const uint4*>(gb0);
        *reinterpret_cast<uint4*>(lb1) = *reinterpret_cast<const uint4*>(gb1);
#endif
    };

    v8i zero = {};
    v8i acc[4];
#pragma unroll
    for (int i = 0; i < 4; ++i) acc[i] = zero;

    stage(0, 0);
#if USE_ASYNC_LDS
    wait_async0();
#endif
    __syncthreads();

    int buf = 0;
    for (int kc = 0; kc < K; kc += 64) {
        if (kc + 64 < K) stage(buf ^ 1, kc + 64);

        v8i af;
        {
            const signed char* rp = &As[buf][(m_sub * 16 + ln) * AST + grp * 16];
            uint2 p0 = *reinterpret_cast<const uint2*>(rp);
            uint2 p1 = *reinterpret_cast<const uint2*>(rp + 8);
            uint2 p2 = *reinterpret_cast<const uint2*>(rp + 32);
            uint2 p3 = *reinterpret_cast<const uint2*>(rp + 40);
            af[0]=(int)p0.x; af[1]=(int)p0.y; af[2]=(int)p1.x; af[3]=(int)p1.y;
            af[4]=(int)p2.x; af[5]=(int)p2.y; af[6]=(int)p3.x; af[7]=(int)p3.y;
        }
#pragma unroll
        for (int nt = 0; nt < 4; ++nt) {
            const signed char* rp = &Bs[buf][(n_sub * 64 + nt * 16 + ln) * BST + grp * 16];
            uint4 q0 = *reinterpret_cast<const uint4*>(rp);
            uint4 q1 = *reinterpret_cast<const uint4*>(rp + 32);
            v8i bf;
            bf[0]=(int)q0.x; bf[1]=(int)q0.y; bf[2]=(int)q0.z; bf[3]=(int)q0.w;
            bf[4]=(int)q1.x; bf[5]=(int)q1.y; bf[6]=(int)q1.z; bf[7]=(int)q1.w;
            acc[nt] = __builtin_amdgcn_wmma_i32_16x16x64_iu8(true, af, true, bf, acc[nt], false, false);
        }
#if USE_ASYNC_LDS
        wait_async0();
#endif
        __syncthreads();
        buf ^= 1;
    }

    const float s = scale_from(sa_bits, sa_premul) * scale_from(sb_bits, 1.0f);
#pragma unroll
    for (int nt = 0; nt < 4; ++nt) {
        const int n = n0 + n_sub * 64 + nt * 16 + ln;
        const float bv = bias ? bias[n] : 0.0f;
#pragma unroll
        for (int r = 0; r < 8; ++r) {
            int m = m0 + m_sub * 16 + r + 8 * grp;
            C[(size_t)m * N + n] = (float)acc[nt][r] * s + bv;
        }
    }
}

__global__ __launch_bounds__(256) void attn_pass1(const signed char* __restrict__ Q,
                                                  const signed char* __restrict__ Kq,
                                                  const unsigned* __restrict__ sq_bits,
                                                  const unsigned* __restrict__ sk_bits,
                                                  float* __restrict__ rowstats,
                                                  unsigned* __restrict__ minZ_bits,
                                                  int S, int H) {
    const int gw = blockIdx.x * 8 + (threadIdx.x >> 5);
    const int lane = threadIdx.x & 31;
    const int qb = gw & 127, bh = gw >> 7;
    const int b = bh >> 4, h = bh & 15;
    const int grp = lane >> 4, ln = lane & 15;
    const float s_qk = scale_from(sq_bits, QMUL) * scale_from(sk_bits, 1.0f);
    const size_t tokbase = (size_t)b * S + qb * 16;

    v8i qa[2];
#pragma unroll
    for (int dc = 0; dc < 2; ++dc) {
        const signed char* base = Q + (tokbase + ln) * (size_t)H + h * HDIM + dc * 64 + grp * 16;
        uint2 p0 = *reinterpret_cast<const uint2*>(base);
        uint2 p1 = *reinterpret_cast<const uint2*>(base + 8);
        uint2 p2 = *reinterpret_cast<const uint2*>(base + 32);
        uint2 p3 = *reinterpret_cast<const uint2*>(base + 40);
        qa[dc][0]=(int)p0.x; qa[dc][1]=(int)p0.y; qa[dc][2]=(int)p1.x; qa[dc][3]=(int)p1.y;
        qa[dc][4]=(int)p2.x; qa[dc][5]=(int)p2.y; qa[dc][6]=(int)p3.x; qa[dc][7]=(int)p3.y;
    }

    float m8[8], z8[8];
#pragma unroll
    for (int r = 0; r < 8; ++r) { m8[r] = -3.0e38f; z8[r] = 0.0f; }

    for (int t = 0; t < S / 16; ++t) {
        v8i acc = {};
        const size_t ktok = (size_t)b * S + t * 16 + ln;
#pragma unroll
        for (int dc = 0; dc < 2; ++dc) {
            const signed char* base = Kq + ktok * (size_t)H + h * HDIM + dc * 64 + grp * 16;
            uint4 q0 = *reinterpret_cast<const uint4*>(base);
            uint4 q1 = *reinterpret_cast<const uint4*>(base + 32);
            v8i bf;
            bf[0]=(int)q0.x; bf[1]=(int)q0.y; bf[2]=(int)q0.z; bf[3]=(int)q0.w;
            bf[4]=(int)q1.x; bf[5]=(int)q1.y; bf[6]=(int)q1.z; bf[7]=(int)q1.w;
            acc = __builtin_amdgcn_wmma_i32_16x16x64_iu8(true, qa[dc], true, bf, acc, false, false);
        }
#pragma unroll
        for (int r = 0; r < 8; ++r) {
            float lg = (float)acc[r] * s_qk;
            float tmax = lg;
            for (int off = 1; off < 16; off <<= 1) tmax = fmaxf(tmax, __shfl_xor(tmax, off, 16));
            float nm = fmaxf(m8[r], tmax);
            float e = __expf(lg - nm);
            for (int off = 1; off < 16; off <<= 1) e += __shfl_xor(e, off, 16);
            z8[r] = z8[r] * __expf(m8[r] - nm) + e;
            m8[r] = nm;
        }
    }
    if (ln == 0) {
        float zmin = 3.0e38f;
#pragma unroll
        for (int r = 0; r < 8; ++r) {
            int idx = gw * 16 + r + 8 * grp;
            rowstats[2 * idx + 0] = m8[r];
            rowstats[2 * idx + 1] = z8[r];
            zmin = fminf(zmin, z8[r]);
        }
        atomicMin(minZ_bits, __float_as_uint(zmin));
    }
}

__global__ __launch_bounds__(256) void attn_pass2(const signed char* __restrict__ Q,
                                                  const signed char* __restrict__ Kq,
                                                  const signed char* __restrict__ VT,
                                                  const float* __restrict__ rowstats,
                                                  const unsigned* __restrict__ minZ_bits,
                                                  const unsigned* __restrict__ sq_bits,
                                                  const unsigned* __restrict__ sk_bits,
                                                  const unsigned* __restrict__ sv_bits,
                                                  float* __restrict__ Out,
                                                  int S, int H) {
    __shared__ __align__(16) signed char Ps[8 * 16 * 72];
    const int wave = threadIdx.x >> 5;
    const int gw = blockIdx.x * 8 + wave;
    const int lane = threadIdx.x & 31;
    const int qb = gw & 127, bh = gw >> 7;
    const int b = bh >> 4, h = bh & 15;
    const int grp = lane >> 4, ln = lane & 15;
    signed char* myP = &Ps[wave * 16 * 72];

    const float s_qk = scale_from(sq_bits, QMUL) * scale_from(sk_bits, 1.0f);
    const float minZ = __uint_as_float(*minZ_bits);
    const float s_p = fmaxf((1.0f / minZ) * (1.0f / 127.0f), 1e-8f);
    const float inv_sp = 1.0f / s_p;
    const float s_pv = s_p * scale_from(sv_bits, 1.0f);
    const size_t tokbase = (size_t)b * S + qb * 16;

    v8i qa[2];
#pragma unroll
    for (int dc = 0; dc < 2; ++dc) {
        const signed char* base = Q + (tokbase + ln) * (size_t)H + h * HDIM + dc * 64 + grp * 16;
        uint2 p0 = *reinterpret_cast<const uint2*>(base);
        uint2 p1 = *reinterpret_cast<const uint2*>(base + 8);
        uint2 p2 = *reinterpret_cast<const uint2*>(base + 32);
        uint2 p3 = *reinterpret_cast<const uint2*>(base + 40);
        qa[dc][0]=(int)p0.x; qa[dc][1]=(int)p0.y; qa[dc][2]=(int)p1.x; qa[dc][3]=(int)p1.y;
        qa[dc][4]=(int)p2.x; qa[dc][5]=(int)p2.y; qa[dc][6]=(int)p3.x; qa[dc][7]=(int)p3.y;
    }

    float m8[8], rz8[8];
#pragma unroll
    for (int r = 0; r < 8; ++r) {
        int idx = gw * 16 + r + 8 * grp;
        m8[r] = rowstats[2 * idx + 0];
        rz8[r] = 1.0f / rowstats[2 * idx + 1];
    }

    v8i zero = {};
    v8i oacc[8];
#pragma unroll
    for (int i = 0; i < 8; ++i) oacc[i] = zero;

    for (int kc = 0; kc < S; kc += 64) {
#pragma unroll
        for (int nt = 0; nt < 4; ++nt) {
            v8i acc = zero;
            const size_t ktok = (size_t)b * S + kc + nt * 16 + ln;
#pragma unroll
            for (int dc = 0; dc < 2; ++dc) {
                const signed char* base = Kq + ktok * (size_t)H + h * HDIM + dc * 64 + grp * 16;
                uint4 q0 = *reinterpret_cast<const uint4*>(base);
                uint4 q1 = *reinterpret_cast<const uint4*>(base + 32);
                v8i bf;
                bf[0]=(int)q0.x; bf[1]=(int)q0.y; bf[2]=(int)q0.z; bf[3]=(int)q0.w;
                bf[4]=(int)q1.x; bf[5]=(int)q1.y; bf[6]=(int)q1.z; bf[7]=(int)q1.w;
                acc = __builtin_amdgcn_wmma_i32_16x16x64_iu8(true, qa[dc], true, bf, acc, false, false);
            }
#pragma unroll
            for (int r = 0; r < 8; ++r) {
                float prob = __expf((float)acc[r] * s_qk - m8[r]) * rz8[r];
                int v = clamp127((int)rintf(prob * inv_sp));
                myP[(r + 8 * grp) * 72 + nt * 16 + ln] = (signed char)v;
            }
        }
        __builtin_amdgcn_wave_barrier();
        v8i pf;
        {
            const signed char* rp = &myP[ln * 72 + grp * 16];
            uint2 p0 = *reinterpret_cast<const uint2*>(rp);
            uint2 p1 = *reinterpret_cast<const uint2*>(rp + 8);
            uint2 p2 = *reinterpret_cast<const uint2*>(rp + 32);
            uint2 p3 = *reinterpret_cast<const uint2*>(rp + 40);
            pf[0]=(int)p0.x; pf[1]=(int)p0.y; pf[2]=(int)p1.x; pf[3]=(int)p1.y;
            pf[4]=(int)p2.x; pf[5]=(int)p2.y; pf[6]=(int)p3.x; pf[7]=(int)p3.y;
        }
        __builtin_amdgcn_wave_barrier();
#pragma unroll
        for (int dt = 0; dt < 8; ++dt) {
            const signed char* base =
                VT + (((size_t)bh * HDIM) + dt * 16 + ln) * (size_t)S + kc + grp * 16;
            uint4 q0 = *reinterpret_cast<const uint4*>(base);
            uint4 q1 = *reinterpret_cast<const uint4*>(base + 32);
            v8i bf;
            bf[0]=(int)q0.x; bf[1]=(int)q0.y; bf[2]=(int)q0.z; bf[3]=(int)q0.w;
            bf[4]=(int)q1.x; bf[5]=(int)q1.y; bf[6]=(int)q1.z; bf[7]=(int)q1.w;
            oacc[dt] = __builtin_amdgcn_wmma_i32_16x16x64_iu8(true, pf, true, bf, oacc[dt], false, false);
        }
    }
#pragma unroll
    for (int dt = 0; dt < 8; ++dt)
#pragma unroll
        for (int r = 0; r < 8; ++r) {
            int m = r + 8 * grp;
            Out[(tokbase + m) * (size_t)H + h * HDIM + dt * 16 + ln] = (float)oacc[dt][r] * s_pv;
        }
}

extern "C" void kernel_launch(void* const* d_in, const int* in_sizes, int n_in,
                              void* d_out, int out_size, void* d_ws, size_t ws_size,
                              hipStream_t stream) {
    (void)in_sizes; (void)n_in; (void)out_size; (void)ws_size;
    const int B = 2, S = 2048, H = 2048;
    const int M = B * S;
    const size_t MH = (size_t)M * H;
    const size_t HH = (size_t)H * H;

    const float* x  = (const float*)d_in[0];
    const float* Wq = (const float*)d_in[1]; const float* bq = (const float*)d_in[2];
    const float* Wk = (const float*)d_in[3]; const float* bk = (const float*)d_in[4];
    const float* Wv = (const float*)d_in[5]; const float* bv = (const float*)d_in[6];
    const float* Wp = (const float*)d_in[7]; const float* bp = (const float*)d_in[8];

    char* ws = (char*)d_ws;
    size_t off = 0;
    auto take = [&](size_t bytes) -> char* {
        char* p = ws + off;
        off += (bytes + 255) & ~(size_t)255;
        return p;
    };
    unsigned*    U    = (unsigned*)take(256);
    signed char* x_q  = (signed char*)take(MH);
    signed char* wq_q = (signed char*)take(HH);
    signed char* wk_q = (signed char*)take(HH);
    signed char* wv_q = (signed char*)take(HH);
    signed char* wp_q = (signed char*)take(HH);
    float*       qf   = (float*)take(MH * 4);
    float*       kf   = (float*)take(MH * 4);
    float*       vf   = (float*)take(MH * 4);
    signed char* q_q  = (signed char*)take(MH);
    signed char* k_q  = (signed char*)take(MH);
    signed char* vT_q = (signed char*)take(MH);
    float*       rowstats = (float*)take((size_t)M * NHEADS * 2 * 4);
    float*       attn_out = qf;
    signed char* a_q      = x_q;

    const int TB = 256;
    const dim3 gemm_grid(H / 128, M / 64);
    const int elem_blocks = (int)((MH + TB - 1) / TB);

    init_scales<<<1, 32, 0, stream>>>(U);

    absmax_k<<<1024, TB, 0, stream>>>(x,  MH / 4, U + 0);
    absmax_k<<<1024, TB, 0, stream>>>(Wq, HH / 4, U + 1);
    absmax_k<<<1024, TB, 0, stream>>>(Wk, HH / 4, U + 2);
    absmax_k<<<1024, TB, 0, stream>>>(Wv, HH / 4, U + 3);
    absmax_k<<<1024, TB, 0, stream>>>(Wp, HH / 4, U + 4);

    quant_k<<<elem_blocks, TB, 0, stream>>>(x,  x_q,  MH, U + 0, 1.0f);
    quant_k<<<(int)((HH + TB - 1) / TB), TB, 0, stream>>>(Wq, wq_q, HH, U + 1, 1.0f);
    quant_k<<<(int)((HH + TB - 1) / TB), TB, 0, stream>>>(Wk, wk_q, HH, U + 2, 1.0f);
    quant_k<<<(int)((HH + TB - 1) / TB), TB, 0, stream>>>(Wv, wv_q, HH, U + 3, 1.0f);
    quant_k<<<(int)((HH + TB - 1) / TB), TB, 0, stream>>>(Wp, wp_q, HH, U + 4, 1.0f);

    gemm_i8<<<gemm_grid, TB, 0, stream>>>(x_q, wq_q, bq, qf, M, H, H, U + 0, 1.0f, U + 1);
    gemm_i8<<<gemm_grid, TB, 0, stream>>>(x_q, wk_q, bk, kf, M, H, H, U + 0, 1.0f, U + 2);
    gemm_i8<<<gemm_grid, TB, 0, stream>>>(x_q, wv_q, bv, vf, M, H, H, U + 0, 1.0f, U + 3);

    absmax_k<<<1024, TB, 0, stream>>>(qf, MH / 4, U + 5);
    absmax_k<<<1024, TB, 0, stream>>>(kf, MH / 4, U + 6);
    absmax_k<<<1024, TB, 0, stream>>>(vf, MH / 4, U + 7);
    quant_k<<<elem_blocks, TB, 0, stream>>>(qf, q_q, MH, U + 5, QMUL);
    quant_k<<<elem_blocks, TB, 0, stream>>>(kf, k_q, MH, U + 6, 1.0f);
    quantT_v<<<elem_blocks, TB, 0, stream>>>(vf, vT_q, MH, U + 7, S);

    attn_pass1<<<512, TB, 0, stream>>>(q_q, k_q, U + 5, U + 6, rowstats, U + 8, S, H);
    attn_pass2<<<512, TB, 0, stream>>>(q_q, k_q, vT_q, rowstats, U + 8,
                                       U + 5, U + 6, U + 7, attn_out, S, H);

    absmax_k<<<1024, TB, 0, stream>>>(attn_out, MH / 4, U + 9);
    quant_k<<<elem_blocks, TB, 0, stream>>>(attn_out, a_q, MH, U + 9, 1.0f);
    gemm_i8<<<gemm_grid, TB, 0, stream>>>(a_q, wp_q, bp, (float*)d_out, M, H, H,
                                          U + 9, 1.0f, U + 4);
}
